// LSTMhead_84026740179152
// MI455X (gfx1250) — compile-verified, matrix operand order checked by reading, not run on hardware
//
#include <hip/hip_runtime.h>

// ---------------------------------------------------------------------------
// MI455X / gfx1250 implementation.
//  - Conv+gather fused: implicit GEMM at sampled points only (16 384 pts).
//  - All matmuls via v_wmma_f32_16x16x32_f16 (wave32, f32 accumulate).
//  - LSTM: one kernel per timestep (serial dep), gate tiles arranged so the
//    cell update is wave-local. Decoder steps fuse the MLP head.
// ---------------------------------------------------------------------------

typedef _Float16 f16;
typedef _Float16 v16h __attribute__((ext_vector_type(16)));
typedef float    v8f  __attribute__((ext_vector_type(8)));

#define NB_   16      // batch
#define NN_   32      // instances
#define NK_   32      // points per instance (= LSTM sequence length)
#define ND_   64      // dot feature dim
#define HID_  128
#define BN_   512     // NB_*NN_
#define NPTS  16384   // NB_*NN_*NK_

__device__ __forceinline__ v8f wmma16(v16h a, v16h b, v8f c) {
  // D = A(16x32 f16) * B(32x16 f16) + C(16x16 f32)
  return __builtin_amdgcn_wmma_f32_16x16x32_f16(false, a, false, b, (short)0, c,
                                                false, false);
}

// Per-lane fragment slice. Caller pre-offsets p to row*ld + kb + ((lane<16)?0:8).
// ISA 16-bit A/B layout: elements 0..7 = K{base..base+7}, 8..15 = K{base+16..+23}.
__device__ __forceinline__ v16h load_frag(const f16* p) {
  v16h r;
#pragma unroll
  for (int i = 0; i < 8; ++i) { r[i] = p[i]; r[i + 8] = p[i + 16]; }
  return r;
}

__device__ __forceinline__ float sigmoidf_(float v) {
  return 1.0f / (1.0f + __expf(-v));
}

// ---------------------------------------------------------------------------
// Prep: fp32 -> f16 weight conversion, bias folding, state zeroing, W2 padding.
// Launched with 65536 threads (covers largest segment).
// ---------------------------------------------------------------------------
__global__ __launch_bounds__(256) void prep_kernel(
    const float* __restrict__ conv1_w, const float* __restrict__ conv2_w,
    const float* __restrict__ WihE, const float* __restrict__ WhhE,
    const float* __restrict__ bihE, const float* __restrict__ bhhE,
    const float* __restrict__ WihD, const float* __restrict__ WhhD,
    const float* __restrict__ bihD, const float* __restrict__ bhhD,
    const float* __restrict__ W1, const float* __restrict__ W2,
    f16* __restrict__ w1c, f16* __restrict__ w2c,
    f16* __restrict__ wihE, f16* __restrict__ whhE,
    f16* __restrict__ wihD, f16* __restrict__ whhD,
    f16* __restrict__ w1m, f16* __restrict__ w2p,
    float* __restrict__ biasE, float* __restrict__ biasD,
    f16* __restrict__ h16, float* __restrict__ cst)
{
  int i = blockIdx.x * blockDim.x + threadIdx.x;   // 0..65535
  if (i < 36864) w1c[i]  = (f16)conv1_w[i];        // [64][576] (c*9+tap contiguous)
  if (i < 4096)  w2c[i]  = (f16)conv2_w[i];        // [64][64]
  if (i < 32768) wihE[i] = (f16)WihE[i];           // [512][64]
  if (i < 65536) whhE[i] = (f16)WhhE[i];           // [512][128]
  if (i < 65536) wihD[i] = (f16)WihD[i];           // [512][128]
  if (i < 65536) whhD[i] = (f16)WhhD[i];           // [512][128]
  if (i < 8192)  w1m[i]  = (f16)W1[i];             // [64][128]
  if (i < 1024)  { int r = i >> 6; w2p[i] = (r < 2) ? (f16)W2[i] : (f16)0.f; } // [16][64]
  if (i < 512)   biasE[i] = bihE[i] + bhhE[i];
  if (i < 512)   biasD[i] = bihD[i] + bhhD[i];
  if (i < 65536) h16[i] = (f16)0.f;                // h0 = 0
  if (i < 65536) cst[i] = 0.f;                     // c0 = 0
}

// ---------------------------------------------------------------------------
// Fused conv3x3 -> ReLU -> conv1x1, evaluated ONLY at the 16384 gathered
// points.  One block = 16 points (one WMMA M-tile), 4 waves = 4 N-tiles of 16
// output channels.  Also emits the gt output.
// ---------------------------------------------------------------------------
__global__ __launch_bounds__(128) void conv_gather_kernel(
    const float* __restrict__ x,           // [16][64][128][128]
    const float* __restrict__ mask_point,  // [16][32][32][2]
    const int*   __restrict__ point_mask,  // [16][32][32]
    const float* __restrict__ conv1_b,     // [64]
    const float* __restrict__ conv2_b,     // [64]
    const f16*   __restrict__ w1c,         // [64][576] f16
    const f16*   __restrict__ w2c,         // [64][64]  f16
    f16*   __restrict__ dots16,            // [16384][64] f16 out
    float* __restrict__ gt_out)            // [16384][2] f32 out
{
  __shared__ f16  patch[16 * 576];   // 18 KB: 16 points x (64ch x 9taps)
  __shared__ f16  feat [16 * 64];    // post-ReLU features
  __shared__ int  pinfo[16 * 3];     // b, y0, x0 per point
  __shared__ float pmaskf[16];

  const int tid = threadIdx.x;
  const int p0  = blockIdx.x * 16;

  if (tid < 16) {
    int p = p0 + tid;
    float px = mask_point[p * 2 + 0] * 0.25f; if (px > 127.f) px = 127.f;
    float py = mask_point[p * 2 + 1] * 0.25f; if (py > 127.f) py = 127.f;
    gt_out[p * 2 + 0] = px * 4.0f;
    gt_out[p * 2 + 1] = py * 4.0f;
    int flat = (int)(py * 128.0f + px);
    pinfo[tid * 3 + 0] = p >> 10;        // b  (N_*K_ = 1024)
    pinfo[tid * 3 + 1] = flat >> 7;      // y0
    pinfo[tid * 3 + 2] = flat & 127;     // x0
    pmaskf[tid] = (float)point_mask[p];
  }
  __syncthreads();

  // Stage the 3x3 x 64ch patches into LDS (zero-padded 'SAME' borders).
  for (int idx = tid; idx < 16 * 576; idx += 128) {
    int pt  = idx / 576, k = idx - pt * 576;
    int ch  = k / 9,   tap = k - ch * 9;
    int dy  = tap / 3 - 1, dx = tap - (tap / 3) * 3 - 1;
    int b   = pinfo[pt * 3 + 0];
    int y   = pinfo[pt * 3 + 1] + dy;
    int xx  = pinfo[pt * 3 + 2] + dx;
    float v = 0.0f;
    if ((unsigned)y < 128u && (unsigned)xx < 128u)
      v = x[(((size_t)b * 64 + ch) * 128 + y) * 128 + xx];
    patch[idx] = (f16)v;
  }
  __syncthreads();

  const int wave   = tid >> 5;
  const int lane   = tid & 31;
  const int colsel = lane & 15;
  const int khalf  = (lane < 16) ? 0 : 8;
  const int rof    = (lane < 16) ? 0 : 8;
  const int n      = wave * 16 + colsel;       // output channel of this lane

  __builtin_prefetch(w1c + (size_t)n * 576, 0, 1);

  // GEMM1: [16 x 576] patch @ [576 x 64] conv1_w^T
  v8f acc = {};
  for (int kb = 0; kb < 576; kb += 32) {
    v16h a  = load_frag(&patch[colsel * 576 + kb + khalf]);
    v16h bf = load_frag(w1c + (size_t)n * 576 + kb + khalf);
    acc = wmma16(a, bf, acc);
  }
  {
    float bb = conv1_b[n];
#pragma unroll
    for (int v = 0; v < 8; ++v) {
      float f = acc[v] + bb;
      feat[(v + rof) * 64 + n] = (f16)(f > 0.f ? f : 0.f);   // ReLU
    }
  }
  __syncthreads();

  // GEMM2: [16 x 64] feat @ [64 x 64] conv2_w^T
  v8f acc2 = {};
  for (int kb = 0; kb < 64; kb += 32) {
    v16h a  = load_frag(&feat[colsel * 64 + kb + khalf]);
    v16h bf = load_frag(w2c + (size_t)n * 64 + kb + khalf);
    acc2 = wmma16(a, bf, acc2);
  }
  {
    float bb = conv2_b[n];
#pragma unroll
    for (int v = 0; v < 8; ++v) {
      int r = v + rof;
      float d = (acc2[v] + bb) * pmaskf[r];                   // point_mask zeroing
      dots16[(size_t)(p0 + r) * 64 + n] = (f16)d;
    }
  }
}

// ---------------------------------------------------------------------------
// One LSTM timestep.  Grid = 32 blocks (16-row M-tiles of the 512 batch rows),
// 256 threads = 8 waves.  Wave w owns gate N-tiles {w, w+8, w+16, w+24} =
// columns {j, 128+j, 256+j, 384+j}, j in [16w,16w+16) -> i/f/g/o for the SAME
// hidden units live in the same lane slots, so the cell update is wave-local.
// Decoder steps (do_mlp) fuse the MLP head and write kp.
// ---------------------------------------------------------------------------
__global__ __launch_bounds__(256) void lstm_step_kernel(
    const f16* __restrict__ xb, int xstride, int Kx,  // x: [512][Kx], row stride
    const f16* __restrict__ wih,    // [512][Kx]  f16
    const f16* __restrict__ whh,    // [512][128] f16
    const float* __restrict__ bias, // [512] (bih+bhh)
    f16*   __restrict__ h16,        // [512][128] f16  in/out
    float* __restrict__ cst,        // [512][128] f32  in/out
    f16*   __restrict__ enc_out,    // nullable: store h for decoder input
    int do_mlp,
    const f16* __restrict__ w1m,    // [64][128] f16
    const float* __restrict__ b1,   // [64]
    const f16* __restrict__ w2p,    // [16][64] f16 (rows 2..15 zero)
    const float* __restrict__ b2,   // [2]
    float* __restrict__ kp_out, int t)
{
  __shared__ f16 lds_h[16 * 128];
  __shared__ f16 lds_z[16 * 64];

  const int tid     = threadIdx.x;
  const int wave    = tid >> 5;
  const int lane    = tid & 31;
  const int colsel  = lane & 15;
  const int khalf   = (lane < 16) ? 0 : 8;
  const int rof     = (lane < 16) ? 0 : 8;
  const int rowbase = blockIdx.x * 16;
  const int j       = wave * 16 + colsel;      // hidden unit owned by this lane

  __builtin_prefetch(whh + (size_t)j * 128, 0, 1);

  v8f acc[4] = {};                             // i, f, g, o gate tiles

  // gates += x @ Wih^T
  for (int kb = 0; kb < Kx; kb += 32) {
    v16h a = load_frag(xb + (size_t)(rowbase + colsel) * xstride + kb + khalf);
#pragma unroll
    for (int q = 0; q < 4; ++q) {
      int n = (wave + 8 * q) * 16 + colsel;    // = q*128 + j
      v16h bf = load_frag(wih + (size_t)n * Kx + kb + khalf);
      acc[q] = wmma16(a, bf, acc[q]);
    }
  }
  // gates += h @ Whh^T
  for (int kb = 0; kb < 128; kb += 32) {
    v16h a = load_frag(h16 + (size_t)(rowbase + colsel) * 128 + kb + khalf);
#pragma unroll
    for (int q = 0; q < 4; ++q) {
      int n = (wave + 8 * q) * 16 + colsel;
      v16h bf = load_frag(whh + (size_t)n * 128 + kb + khalf);
      acc[q] = wmma16(a, bf, acc[q]);
    }
  }
  __syncthreads();   // all waves finished reading h16 before anyone rewrites it

  // Cell update (PyTorch gate order i, f, g, o).
  const float bi = bias[j],       bfv = bias[128 + j];
  const float bg = bias[256 + j], bo  = bias[384 + j];
  float hloc[8];
#pragma unroll
  for (int v = 0; v < 8; ++v) {
    int r = rowbase + v + rof;
    float gi = acc[0][v] + bi;
    float gf = acc[1][v] + bfv;
    float gg = acc[2][v] + bg;
    float go = acc[3][v] + bo;
    float cn = sigmoidf_(gf) * cst[(size_t)r * 128 + j] +
               sigmoidf_(gi) * tanhf(gg);
    cst[(size_t)r * 128 + j] = cn;
    float hn = sigmoidf_(go) * tanhf(cn);
    h16[(size_t)r * 128 + j] = (f16)hn;
    if (enc_out) enc_out[(size_t)r * 128 + j] = (f16)hn;
    hloc[v] = hn;
  }

  if (do_mlp) {
    // Stage this M-tile's h into LDS for the MLP GEMMs.
#pragma unroll
    for (int v = 0; v < 8; ++v)
      lds_h[(v + rof) * 128 + j] = (f16)hloc[v];
    __syncthreads();

    if (wave < 4) {                       // z = relu(h @ W1^T + b1): N=64
      v8f z = {};
      int n = wave * 16 + colsel;
      for (int kb = 0; kb < 128; kb += 32) {
        v16h a  = load_frag(&lds_h[colsel * 128 + kb + khalf]);
        v16h bf = load_frag(w1m + (size_t)n * 128 + kb + khalf);
        z = wmma16(a, bf, z);
      }
      float bb = b1[n];
#pragma unroll
      for (int v = 0; v < 8; ++v) {
        float zv = z[v] + bb;
        lds_z[(v + rof) * 64 + n] = (f16)(zv > 0.f ? zv : 0.f);
      }
    }
    __syncthreads();

    if (wave == 0) {                      // y = z @ W2pad^T: cols 0,1 valid
      v8f y = {};
      for (int kb = 0; kb < 64; kb += 32) {
        v16h a  = load_frag(&lds_z[colsel * 64 + kb + khalf]);
        v16h bf = load_frag(w2p + (size_t)colsel * 64 + kb + khalf);
        y = wmma16(a, bf, y);
      }
      if (colsel < 2) {
        float bb = b2[colsel];
#pragma unroll
        for (int v = 0; v < 8; ++v) {
          int r = rowbase + v + rof;
          // kp[b][n][k][c] = y[t=k][row=m][c] -> flat (m*32 + t)*2 + c
          kp_out[((size_t)r * NK_ + t) * 2 + colsel] = y[v] + bb;
        }
      }
    }
  }
}

// ---------------------------------------------------------------------------
extern "C" void kernel_launch(void* const* d_in, const int* in_sizes, int n_in,
                              void* d_out, int out_size, void* d_ws, size_t ws_size,
                              hipStream_t stream) {
  (void)in_sizes; (void)n_in; (void)out_size; (void)ws_size;

  const float* x          = (const float*)d_in[0];
  const float* mask_point = (const float*)d_in[1];
  const int*   point_mask = (const int*)  d_in[2];
  const float* conv1_w    = (const float*)d_in[3];
  const float* conv1_b    = (const float*)d_in[4];
  const float* conv2_w    = (const float*)d_in[5];
  const float* conv2_b    = (const float*)d_in[6];
  const float* WihE = (const float*)d_in[7];
  const float* WhhE = (const float*)d_in[8];
  const float* bihE = (const float*)d_in[9];
  const float* bhhE = (const float*)d_in[10];
  const float* WihD = (const float*)d_in[11];
  const float* WhhD = (const float*)d_in[12];
  const float* bihD = (const float*)d_in[13];
  const float* bhhD = (const float*)d_in[14];
  const float* W1   = (const float*)d_in[15];
  const float* b1   = (const float*)d_in[16];
  const float* W2   = (const float*)d_in[17];
  const float* b2   = (const float*)d_in[18];

  float* out    = (float*)d_out;          // kp [16384][2]
  float* gt_out = out + NPTS * 2;         // gt [16384][2]

  // Workspace carve-out (~7.25 MB total), 256B aligned.
  char* w = (char*)d_ws;
  auto alloc = [&](size_t bytes) {
    char* p = w; w += (bytes + 255) & ~(size_t)255; return p;
  };
  f16*   w1c    = (f16*)  alloc(36864 * 2);          // conv1_w f16 [64][576]
  f16*   w2c    = (f16*)  alloc(4096 * 2);           // conv2_w f16 [64][64]
  f16*   wihE16 = (f16*)  alloc(32768 * 2);          // [512][64]
  f16*   whhE16 = (f16*)  alloc(65536 * 2);          // [512][128]
  f16*   wihD16 = (f16*)  alloc(65536 * 2);          // [512][128]
  f16*   whhD16 = (f16*)  alloc(65536 * 2);          // [512][128]
  f16*   w1m16  = (f16*)  alloc(8192 * 2);           // [64][128]
  f16*   w2p16  = (f16*)  alloc(1024 * 2);           // [16][64] padded
  float* biasE  = (float*)alloc(512 * 4);
  float* biasD  = (float*)alloc(512 * 4);
  f16*   h16    = (f16*)  alloc(65536 * 2);          // [512][128]
  float* cst    = (float*)alloc(65536 * 4);          // [512][128]
  f16*   dots16 = (f16*)  alloc((size_t)NPTS * ND_ * 2);       // [16384][64]
  f16*   ench16 = (f16*)  alloc((size_t)NK_ * BN_ * HID_ * 2); // [32][512][128]

  // 1) weight conversion + state init (re-done every call: no cached state).
  prep_kernel<<<256, 256, 0, stream>>>(
      conv1_w, conv2_w, WihE, WhhE, bihE, bhhE, WihD, WhhD, bihD, bhhD, W1, W2,
      w1c, w2c, wihE16, whhE16, wihD16, whhD16, w1m16, w2p16,
      biasE, biasD, h16, cst);

  // 2) fused conv+gather at the 16384 sample points (+ gt output).
  conv_gather_kernel<<<NPTS / 16, 128, 0, stream>>>(
      x, mask_point, point_mask, conv1_b, conv2_b, w1c, w2c, dots16, gt_out);

  // 3) encoder: 32 serial steps; x_t = dots[:, t, :] (row stride 32*64).
  for (int t = 0; t < NK_; ++t)
    lstm_step_kernel<<<BN_ / 16, 256, 0, stream>>>(
        dots16 + (size_t)t * ND_, NK_ * ND_, ND_,
        wihE16, whhE16, biasE, h16, cst,
        ench16 + (size_t)t * BN_ * HID_,
        0, nullptr, nullptr, nullptr, nullptr, out, t);

  // 4) decoder (+ fused MLP head): (h,c) carried over from the encoder.
  for (int t = 0; t < NK_; ++t)
    lstm_step_kernel<<<BN_ / 16, 256, 0, stream>>>(
        ench16 + (size_t)t * BN_ * HID_, HID_, HID_,
        wihD16, whhD16, biasD, h16, cst,
        nullptr,
        1, w1m16, b1, w2p16, b2, out, t);
}
